// NystromAttention_11287174054622
// MI455X (gfx1250) — matrix operand order repaired and checked by reading, not compiled, not run on hardware
//
#include <hip/hip_runtime.h>
#include <math.h>

// ---------------------------------------------------------------------------
// Nystrom attention for MI455X (gfx1250), wave32, WMMA bf16 path.
//   [N=4, L=8192, H=16, D=64], m=64 landmarks, seg=128, fp32 io.
// Memory-bound (~0.77GB traffic @ 23.3TB/s ~ 33us); all large GEMMs use
// v_wmma_f32_16x16x32_bf16 (fp32 accumulate); softmax & Newton-Schulz in fp32.
// All WMMA operands staged K-contiguous in LDS -> ds_load_b128 fragments.
// ---------------------------------------------------------------------------

typedef __attribute__((ext_vector_type(16))) __bf16 bf16x16;
typedef __attribute__((ext_vector_type(8)))  __bf16 bf16x8;
typedef __attribute__((ext_vector_type(8)))  float  f32x8;

#define N_   4
#define L_   8192
#define H_   16
#define D_   64
#define M_   64
#define SEG_ 128
#define NH_  64
#define P_   8            // split-K partitions over L for stage 3
#define LP_  (L_ / P_)    // 1024 columns per partition
#define CH_  64           // chunk of L per inner iteration
#define SCALE_F 0.35355339059327373f   // 64^(-0.25)

// ----------------------------- WMMA helpers --------------------------------
__device__ __forceinline__ bf16x16 cat8(bf16x8 lo, bf16x8 hi) {
  union { bf16x16 v; bf16x8 h[2]; } u;
  u.h[0] = lo; u.h[1] = hi;
  return u.v;
}

// A fragment: 16x32 bf16 tile, row-major LDS, row stride lda (multiple of 8).
// Per ISA 7.12.2: lane half=lane>>4, row=lane&15;
//   vgprs 0..3 hold K = k0+half*8 + 0..7, vgprs 4..7 hold K = k0+16+half*8+0..7
//   -> two contiguous 16B runs -> 2x ds_load_b128.
__device__ __forceinline__ bf16x16 load_a_frag(const __bf16* A, int lda,
                                               int m0, int k0, int lane) {
  const int half = lane >> 4, r = lane & 15;
  const __bf16* p = A + (m0 + r) * lda + k0 + half * 8;
  return cat8(*(const bf16x8*)p, *(const bf16x8*)(p + 16));
}

// B fragment: 32x16 bf16 tile; operand stored n-major, K-contiguous:
//   element(k,n) = B[n*ldn + k].  Lane n=lane&15, half=lane>>4;
//   vgprs 0..3 hold K = k0+half*8 + 0..7, vgprs 4..7 hold K = k0+16+half*8+0..7
__device__ __forceinline__ bf16x16 load_b_frag(const __bf16* B, int ldn,
                                               int k0, int n0, int lane) {
  const int half = lane >> 4, n = lane & 15;
  const __bf16* p = B + (n0 + n) * ldn + k0 + half * 8;
  return cat8(*(const bf16x8*)p, *(const bf16x8*)(p + 16));
}

// C/D 16x16 f32: vgpr i -> row = half*8 + i, col = lane&15.
__device__ __forceinline__ void store_c_tile(float* S, int lds_, int m0, int n0,
                                             const f32x8& c, int lane) {
  const int half = lane >> 4, col = lane & 15;
#pragma unroll
  for (int i = 0; i < 8; ++i)
    S[(m0 + half * 8 + i) * lds_ + n0 + col] = c[i];
}

__device__ __forceinline__ f32x8 wmma_bf16(bf16x16 a, bf16x16 b, f32x8 c) {
  return __builtin_amdgcn_wmma_f32_16x16x32_bf16(
      /*neg_a=*/false, a, /*neg_b=*/false, b,
      /*c_mod=*/(short)0, c, /*reuse_a=*/false, /*reuse_b=*/false);
}

// pack two float4 into one bf16x8 (optionally scaled)
__device__ __forceinline__ bf16x8 pack8(float4 a, float4 b, float s) {
  bf16x8 r;
  r[0] = (__bf16)(a.x * s); r[1] = (__bf16)(a.y * s);
  r[2] = (__bf16)(a.z * s); r[3] = (__bf16)(a.w * s);
  r[4] = (__bf16)(b.x * s); r[5] = (__bf16)(b.y * s);
  r[6] = (__bf16)(b.z * s); r[7] = (__bf16)(b.w * s);
  return r;
}

// ---------------------- Kernel 1: landmark pooling -------------------------
__global__ __launch_bounds__(128) void k_landmarks(
    const float* __restrict__ q, const float* __restrict__ kk,
    float* __restrict__ ql_f, float* __restrict__ kl_f,
    __bf16* __restrict__ ql_b, __bf16* __restrict__ kl_b) {
  const int b = blockIdx.x;
  const int n = b / (M_ * H_);
  const int lm = (b / H_) % M_;
  const int h = b % H_;
  const int t = threadIdx.x;
  const int d = t & 63;
  const float* src = (t < 64) ? q : kk;
  size_t base = ((size_t)(n * L_ + lm * SEG_) * H_ + h) * D_ + d;
  float s = 0.f;
  for (int r = 0; r < SEG_; ++r) s += src[base + (size_t)r * H_ * D_];
  s = s * (SCALE_F / (float)SEG_);
  const size_t o = ((size_t)((n * H_ + h) * M_ + lm)) * D_ + d;
  if (t < 64) { ql_f[o] = s; ql_b[o] = (__bf16)s; }
  else        { kl_f[o] = s; kl_b[o] = (__bf16)s; }
}

// -------- Kernel 2: kernel_2 softmax + Newton-Schulz inverse (fp32) --------
__device__ __forceinline__ void mm64(float* D, const float* A, const float* B,
                                     int t) {
  for (int e = 0; e < 16; ++e) {
    const int idx = t * 16 + e, i = idx >> 6, j = idx & 63;
    float s = 0.f;
#pragma unroll 8
    for (int d = 0; d < 64; ++d) s += A[i * 64 + d] * B[d * 64 + j];
    D[idx] = s;
  }
}

__global__ __launch_bounds__(256) void k_kernel2_inv(
    const float* __restrict__ ql_f, const float* __restrict__ kl_f,
    float* __restrict__ Kg, float* __restrict__ inv_f) {
  __shared__ float Vb[64 * 64];
  __shared__ float KVb[64 * 64];
  __shared__ float Tb[64 * 64 + 64];
  __shared__ float Ub[64 * 64];
  const int nh = blockIdx.x, t = threadIdx.x;
  const size_t gb = (size_t)nh * 4096;
  for (int e = 0; e < 16; ++e) {
    const int idx = e * 256 + t;
    KVb[idx] = ql_f[gb + idx];  // q_l staged
    Tb[idx]  = kl_f[gb + idx];  // k_l staged
  }
  __syncthreads();
  // S = q_l @ k_l^T  -> Ub
  for (int e = 0; e < 16; ++e) {
    const int idx = t * 16 + e, i = idx >> 6, j = idx & 63;
    float s = 0.f;
#pragma unroll 8
    for (int d = 0; d < 64; ++d) s += KVb[i * 64 + d] * Tb[j * 64 + d];
    Ub[idx] = s;
  }
  __syncthreads();
  if (t < 64) {  // row softmax
    float mx = -1e30f;
    for (int j = 0; j < 64; ++j) mx = fmaxf(mx, Ub[t * 64 + j]);
    float sm = 0.f;
    for (int j = 0; j < 64; ++j) {
      float e2 = expf(Ub[t * 64 + j] - mx);
      Ub[t * 64 + j] = e2;
      sm += e2;
    }
    const float inv = 1.f / sm;
    for (int j = 0; j < 64; ++j) Ub[t * 64 + j] *= inv;
  }
  __syncthreads();
  if (t < 64) {  // column sums
    float s = 0.f;
    for (int i = 0; i < 64; ++i) s += Ub[i * 64 + t];
    Tb[t] = s;
  }
  __syncthreads();
  if (t == 0) {
    float mx = Tb[0];
    for (int j = 1; j < 64; ++j) mx = fmaxf(mx, Tb[j]);
    Tb[64 * 64] = mx;
  }
  __syncthreads();
  const float denom = Tb[64 * 64];
  for (int e = 0; e < 16; ++e) {
    const int idx = t * 16 + e, i = idx >> 6, j = idx & 63;
    Vb[idx] = Ub[j * 64 + i] / denom;  // V0 = K^T / denom
    Kg[gb + idx] = Ub[idx];            // persist K (L2-resident scratch)
  }
  __syncthreads();
  for (int it = 0; it < 6; ++it) {
    for (int e = 0; e < 16; ++e) {  // KV = K @ V
      const int idx = t * 16 + e, i = idx >> 6, j = idx & 63;
      const float* Krow = Kg + gb + i * 64;
      float s = 0.f;
#pragma unroll 8
      for (int d = 0; d < 64; ++d) s += Krow[d] * Vb[d * 64 + j];
      KVb[idx] = s;
    }
    __syncthreads();
    for (int e = 0; e < 16; ++e) {
      const int idx = t * 16 + e, i = idx >> 6, j = idx & 63;
      Tb[idx] = (i == j ? 7.f : 0.f) - KVb[idx];
    }
    __syncthreads();
    mm64(Ub, KVb, Tb, t);
    __syncthreads();
    for (int e = 0; e < 16; ++e) {
      const int idx = t * 16 + e, i = idx >> 6, j = idx & 63;
      Tb[idx] = (i == j ? 15.f : 0.f) - Ub[idx];
    }
    __syncthreads();
    mm64(Ub, KVb, Tb, t);
    __syncthreads();
    for (int e = 0; e < 16; ++e) {
      const int idx = t * 16 + e, i = idx >> 6, j = idx & 63;
      Tb[idx] = (i == j ? 13.f : 0.f) - Ub[idx];
    }
    __syncthreads();
    mm64(Ub, Vb, Tb, t);
    __syncthreads();
    for (int e = 0; e < 16; ++e) {
      const int idx = t * 16 + e;
      Vb[idx] = 0.25f * Ub[idx];
    }
    __syncthreads();
  }
  for (int e = 0; e < 16; ++e) {
    const int idx = e * 256 + t;
    inv_f[gb + idx] = Vb[idx];
  }
}

// ------ Kernel 3: split-K flash, tmp_p = sm(q_l k^T) v partials (WMMA) -----
__global__ __launch_bounds__(256) void k_flash_part(
    const float* __restrict__ kg, const float* __restrict__ vg,
    const __bf16* __restrict__ ql_b, float* __restrict__ part_o,
    float* __restrict__ part_m, float* __restrict__ part_s) {
  __shared__ __bf16 QL[64 * 64];    // q_l, row-major [m][d] (A operand)
  __shared__ __bf16 Kc[64 * 64];    // k chunk, row-major [l][d] (B: k=d contig)
  __shared__ __bf16 VcT[64 * 64];   // v chunk TRANSPOSED [d][l] (B: k=l contig)
  __shared__ float  Sc[64 * 64];
  __shared__ __bf16 Pc[64 * 64];    // probs, row-major [m][l] (A operand)
  __shared__ float rowm[64], rows[64], alpha[64];
  const int p = blockIdx.x, nh = blockIdx.y;
  const int n = nh >> 4, h = nh & 15;
  const int t = threadIdx.x, w = t >> 5, lane = t & 31;
  {  // q_l LDS copy via uint4 (8 bf16 per uint4)
    const uint4* src = (const uint4*)(ql_b + (size_t)nh * 4096);
    uint4* dst = (uint4*)QL;
#pragma unroll
    for (int e = 0; e < 2; ++e) dst[e * 256 + t] = src[e * 256 + t];
  }
  if (t < 64) { rowm[t] = -1e30f; rows[t] = 0.f; }
  const int id0 = w * 2, id1 = w * 2 + 1;
  const int mt0 = id0 >> 2, nt0 = id0 & 3, mt1 = id1 >> 2, nt1 = id1 & 3;
  f32x8 acc0 = {}; f32x8 acc1 = {};
  __syncthreads();
  for (int c = 0; c < LP_ / CH_; ++c) {
    const int l0 = p * LP_ + c * CH_;
    {  // stage k chunk (scaled) row-major: 2x ds_store_b128 per thread
      const int r = t >> 2, d0 = (t & 3) * 16;
      const size_t gidx = ((size_t)(n * L_ + l0 + r) * H_ + h) * D_ + d0;
      const float4* kp = (const float4*)(kg + gidx);
      *(bf16x8*)(Kc + r * 64 + d0)     = pack8(kp[0], kp[1], SCALE_F);
      *(bf16x8*)(Kc + r * 64 + d0 + 8) = pack8(kp[2], kp[3], SCALE_F);
      if (c + 1 < LP_ / CH_) {  // prefetch next chunk (global_prefetch_b8)
        __builtin_prefetch(kg + gidx + (size_t)CH_ * H_ * D_, 0, 1);
        __builtin_prefetch(vg + gidx + (size_t)CH_ * H_ * D_, 0, 1);
      }
    }
    {  // stage v chunk transposed [d][l]: thread owns dim dv, 16 rows
      const int dv = t & 63, lg = (t >> 6) * 16;
      const float* vb =
          vg + ((size_t)(n * L_ + l0 + lg) * H_ + h) * D_ + dv;
      bf16x8 va, vbp;
#pragma unroll
      for (int j = 0; j < 8; ++j)
        va[j] = (__bf16)vb[(size_t)j * H_ * D_];
#pragma unroll
      for (int j = 0; j < 8; ++j)
        vbp[j] = (__bf16)vb[(size_t)(j + 8) * H_ * D_];
      *(bf16x8*)(VcT + dv * 64 + lg)     = va;
      *(bf16x8*)(VcT + dv * 64 + lg + 8) = vbp;
    }
    __syncthreads();
    {  // S = q_l (64x64) @ Kc^T -> Sc
      f32x8 s0 = {}; f32x8 s1 = {};
#pragma unroll
      for (int k0 = 0; k0 < 64; k0 += 32) {
        s0 = wmma_bf16(load_a_frag(QL, 64, mt0 * 16, k0, lane),
                       load_b_frag(Kc, 64, k0, nt0 * 16, lane), s0);
        s1 = wmma_bf16(load_a_frag(QL, 64, mt1 * 16, k0, lane),
                       load_b_frag(Kc, 64, k0, nt1 * 16, lane), s1);
      }
      store_c_tile(Sc, 64, mt0 * 16, nt0 * 16, s0, lane);
      store_c_tile(Sc, 64, mt1 * 16, nt1 * 16, s1, lane);
    }
    __syncthreads();
    if (t < 64) {  // online softmax update (fp32), packed bf16 prob writes
      float m_new = rowm[t];
      for (int j = 0; j < 64; ++j) m_new = fmaxf(m_new, Sc[t * 64 + j]);
      const float a = expf(rowm[t] - m_new);
      float s = rows[t] * a;
      for (int jj = 0; jj < 64; jj += 8) {
        bf16x8 pk;
#pragma unroll
        for (int j2 = 0; j2 < 8; ++j2) {
          const float e2 = expf(Sc[t * 64 + jj + j2] - m_new);
          s += e2;
          pk[j2] = (__bf16)e2;
        }
        *(bf16x8*)(Pc + t * 64 + jj) = pk;
      }
      rowm[t] = m_new; rows[t] = s; alpha[t] = a;
    }
    __syncthreads();
    {  // O = alpha*O + P @ V
      const int half = lane >> 4;
#pragma unroll
      for (int i = 0; i < 8; ++i) {
        acc0[i] *= alpha[mt0 * 16 + half * 8 + i];
        acc1[i] *= alpha[mt1 * 16 + half * 8 + i];
      }
#pragma unroll
      for (int k0 = 0; k0 < 64; k0 += 32) {
        acc0 = wmma_bf16(load_a_frag(Pc, 64, mt0 * 16, k0, lane),
                         load_b_frag(VcT, 64, k0, nt0 * 16, lane), acc0);
        acc1 = wmma_bf16(load_a_frag(Pc, 64, mt1 * 16, k0, lane),
                         load_b_frag(VcT, 64, k0, nt1 * 16, lane), acc1);
      }
    }
    __syncthreads();
  }
  // write partial (unnormalized O, row max, row sum)
  store_c_tile(Sc, 64, mt0 * 16, nt0 * 16, acc0, lane);
  store_c_tile(Sc, 64, mt1 * 16, nt1 * 16, acc1, lane);
  __syncthreads();
  const size_t ob = ((size_t)nh * P_ + p) * 4096;
  for (int e = 0; e < 16; ++e) {
    const int idx = e * 256 + t;
    part_o[ob + idx] = Sc[idx];
  }
  if (t < 64) {
    part_m[((size_t)nh * P_ + p) * 64 + t] = rowm[t];
    part_s[((size_t)nh * P_ + p) * 64 + t] = rows[t];
  }
}

// --------- Kernel 4: combine partials -> tmp, then W = inv2 @ tmp ----------
__global__ __launch_bounds__(256) void k_combine_w(
    const float* __restrict__ part_o, const float* __restrict__ part_m,
    const float* __restrict__ part_s, const float* __restrict__ inv_f,
    __bf16* __restrict__ W_b) {
  __shared__ float rowm4[64], ltot[64];
  __shared__ float Tt[4096];
  __shared__ float Iv[4096];
  const int nh = blockIdx.x, t = threadIdx.x;
  for (int e = 0; e < 16; ++e) {
    const int idx = e * 256 + t;
    Iv[idx] = inv_f[(size_t)nh * 4096 + idx];
  }
  if (t < 64) {
    float m = -1e30f;
    for (int p = 0; p < P_; ++p)
      m = fmaxf(m, part_m[((size_t)nh * P_ + p) * 64 + t]);
    float l = 0.f;
    for (int p = 0; p < P_; ++p)
      l += part_s[((size_t)nh * P_ + p) * 64 + t] *
           expf(part_m[((size_t)nh * P_ + p) * 64 + t] - m);
    rowm4[t] = m; ltot[t] = l;
  }
  __syncthreads();
  for (int e = 0; e < 16; ++e) {
    const int idx = t * 16 + e, i = idx >> 6;
    float o = 0.f;
    for (int p = 0; p < P_; ++p)
      o += part_o[((size_t)nh * P_ + p) * 4096 + idx] *
           expf(part_m[((size_t)nh * P_ + p) * 64 + i] - rowm4[i]);
    Tt[idx] = o / ltot[i];  // tmp[m][d]
  }
  __syncthreads();
  for (int e = 0; e < 16; ++e) {  // W = inv2 @ tmp
    const int idx = t * 16 + e, i = idx >> 6, j = idx & 63;
    float s = 0.f;
#pragma unroll 8
    for (int d = 0; d < 64; ++d) s += Iv[i * 64 + d] * Tt[d * 64 + j];
    W_b[(size_t)nh * 4096 + idx] = (__bf16)s;
  }
}

// ---------- Kernel 5: out = softmax(q @ k_l^T) @ W (WMMA, tiled L) ---------
__global__ __launch_bounds__(256) void k_final(
    const float* __restrict__ qg, const __bf16* __restrict__ kl_b,
    const __bf16* __restrict__ W_b, float* __restrict__ out) {
  __shared__ __bf16 Qc[64 * 64];    // q rows, row-major [l][d] (A operand)
  __shared__ __bf16 KL[64 * 64];    // k_l row-major [m][d] (B: k=d contig)
  __shared__ __bf16 WlT[64 * 64];   // W TRANSPOSED [d][m] (B: k=m contig)
  __shared__ float  Sc[64 * 64];
  __shared__ __bf16 Pc[64 * 64];    // probs row-major [l][m] (A operand)
  const int bx = blockIdx.x, nh = blockIdx.y;
  const int n = nh >> 4, h = nh & 15;
  const int t = threadIdx.x, w = t >> 5, lane = t & 31;
  const int l0 = bx * 64;
  {  // k_l: straight uint4 copy; W: transpose on store
    const uint4* src = (const uint4*)(kl_b + (size_t)nh * 4096);
    uint4* dst = (uint4*)KL;
#pragma unroll
    for (int e = 0; e < 2; ++e) dst[e * 256 + t] = src[e * 256 + t];
    const __bf16* wsrc = W_b + (size_t)nh * 4096;
#pragma unroll
    for (int e = 0; e < 16; ++e) {
      const int idx = e * 256 + t;
      WlT[(idx & 63) * 64 + (idx >> 6)] = wsrc[idx];
    }
  }
  {  // stage scaled q rows -> bf16 LDS, 2x ds_store_b128 per thread
    const int r = t >> 2, d0 = (t & 3) * 16;
    const size_t gidx = ((size_t)(n * L_ + l0 + r) * H_ + h) * D_ + d0;
    const float4* qp = (const float4*)(qg + gidx);
    *(bf16x8*)(Qc + r * 64 + d0)     = pack8(qp[0], qp[1], SCALE_F);
    *(bf16x8*)(Qc + r * 64 + d0 + 8) = pack8(qp[2], qp[3], SCALE_F);
  }
  __syncthreads();
  const int id0 = w * 2, id1 = w * 2 + 1;
  const int mt0 = id0 >> 2, nt0 = id0 & 3, mt1 = id1 >> 2, nt1 = id1 & 3;
  f32x8 s0 = {}; f32x8 s1 = {};
#pragma unroll
  for (int k0 = 0; k0 < 64; k0 += 32) {  // S = Q @ k_l^T
    s0 = wmma_bf16(load_a_frag(Qc, 64, mt0 * 16, k0, lane),
                   load_b_frag(KL, 64, k0, nt0 * 16, lane), s0);
    s1 = wmma_bf16(load_a_frag(Qc, 64, mt1 * 16, k0, lane),
                   load_b_frag(KL, 64, k0, nt1 * 16, lane), s1);
  }
  store_c_tile(Sc, 64, mt0 * 16, nt0 * 16, s0, lane);
  store_c_tile(Sc, 64, mt1 * 16, nt1 * 16, s1, lane);
  __syncthreads();
  if (t < 64) {  // exact row softmax over 64 landmarks (fp32)
    float mx = -1e30f;
    for (int j = 0; j < 64; ++j) mx = fmaxf(mx, Sc[t * 64 + j]);
    float sm = 0.f;
    for (int j = 0; j < 64; ++j) {
      const float e2 = expf(Sc[t * 64 + j] - mx);
      Sc[t * 64 + j] = e2;
      sm += e2;
    }
    const float inv = 1.f / sm;
    for (int jj = 0; jj < 64; jj += 8) {
      bf16x8 pk;
#pragma unroll
      for (int j2 = 0; j2 < 8; ++j2)
        pk[j2] = (__bf16)(Sc[t * 64 + jj + j2] * inv);
      *(bf16x8*)(Pc + t * 64 + jj) = pk;
    }
  }
  __syncthreads();
  f32x8 o0 = {}; f32x8 o1 = {};
#pragma unroll
  for (int k0 = 0; k0 < 64; k0 += 32) {  // Out = P @ W
    o0 = wmma_bf16(load_a_frag(Pc, 64, mt0 * 16, k0, lane),
                   load_b_frag(WlT, 64, k0, nt0 * 16, lane), o0);
    o1 = wmma_bf16(load_a_frag(Pc, 64, mt1 * 16, k0, lane),
                   load_b_frag(WlT, 64, k0, nt1 * 16, lane), o1);
  }
  store_c_tile(Sc, 64, mt0 * 16, nt0 * 16, o0, lane);
  store_c_tile(Sc, 64, mt1 * 16, nt1 * 16, o1, lane);
  __syncthreads();
  {  // coalesced fp32 writes through LDS staging
    const int r = t >> 2, d0 = (t & 3) * 16;
    const size_t gidx = ((size_t)(n * L_ + l0 + r) * H_ + h) * D_ + d0;
    float4* op = (float4*)(out + gidx);
    const float4* sp = (const float4*)(Sc + r * 64 + d0);
#pragma unroll
    for (int x = 0; x < 4; ++x) op[x] = sp[x];
  }
}

// ------------------------------- launcher ----------------------------------
extern "C" void kernel_launch(void* const* d_in, const int* in_sizes, int n_in,
                              void* d_out, int out_size, void* d_ws,
                              size_t ws_size, hipStream_t stream) {
  (void)in_sizes; (void)n_in; (void)out_size; (void)ws_size;
  const float* q = (const float*)d_in[0];
  const float* k = (const float*)d_in[1];
  const float* v = (const float*)d_in[2];
  float* out = (float*)d_out;
  char* ws = (char*)d_ws;
  // workspace layout (~13.75 MB total)
  float*  ql_f   = (float*)(ws);
  float*  kl_f   = (float*)(ws + (1u << 20));
  __bf16* ql_b   = (__bf16*)(ws + (2u << 20));
  __bf16* kl_b   = (__bf16*)(ws + (2u << 20) + (512u << 10));
  float*  inv_f  = (float*)(ws + (3u << 20));
  float*  Kg     = (float*)(ws + (4u << 20));
  float*  part_o = (float*)(ws + (5u << 20));
  float*  part_m = (float*)(ws + (13u << 20));
  float*  part_s = (float*)(ws + (13u << 20) + (128u << 10));
  __bf16* W_b    = (__bf16*)(ws + (13u << 20) + (256u << 10));

  k_landmarks<<<dim3(N_ * M_ * H_), dim3(128), 0, stream>>>(
      q, k, ql_f, kl_f, ql_b, kl_b);
  k_kernel2_inv<<<dim3(NH_), dim3(256), 0, stream>>>(ql_f, kl_f, Kg, inv_f);
  k_flash_part<<<dim3(P_, NH_), dim3(256), 0, stream>>>(
      k, v, ql_b, part_o, part_m, part_s);
  k_combine_w<<<dim3(NH_), dim3(256), 0, stream>>>(
      part_o, part_m, part_s, inv_f, W_b);
  k_final<<<dim3(L_ / 64, NH_), dim3(256), 0, stream>>>(q, kl_b, W_b, out);
}
